// MultiScaleSpatialAttention_13314398617806
// MI455X (gfx1250) — matrix operand order repaired and checked by reading, not compiled, not run on hardware
//
#include <hip/hip_runtime.h>

#define NN   50000
#define DD   128
#define HH   8
#define EE   600000
#define NSC  3
#define NBIN 50

typedef __attribute__((ext_vector_type(16))) __bf16 v16bf;
typedef __attribute__((ext_vector_type(8)))  float  v8f;

// ---------- helpers ----------
__device__ __forceinline__ unsigned short f2bfbits(float f) {
  unsigned int u = __float_as_uint(f);
  u += 0x7fffu + ((u >> 16) & 1u);        // round-to-nearest-even
  return (unsigned short)(u >> 16);
}

__device__ __forceinline__ void atomicMaxF(float* addr, float val) {
  if (val >= 0.0f) atomicMax((int*)addr, __float_as_int(val));
  else             atomicMin((unsigned int*)addr, __float_as_uint(val));
}

// ---------- elementwise utility kernels ----------
__global__ void conv_bf16_kernel(const float* __restrict__ src,
                                 unsigned short* __restrict__ dst, int n) {
  int i = blockIdx.x * blockDim.x + threadIdx.x;
  if (i < n) dst[i] = f2bfbits(src[i]);
}

__global__ void fill_kernel(float* __restrict__ p, int n, float val) {
  int i = blockIdx.x * blockDim.x + threadIdx.x;
  if (i < n) p[i] = val;
}

// ---------- WMMA bf16 GEMM: C[m,j] = sum_k A[m,k]*W[j,k] + bias[j] ----------
// A: NN x 128 bf16 row-major, W: 128 x 128 bf16 row-major, C: NN x 128 f32.
// One wave computes a 16x64 slab (4 column tiles), reusing the A fragment 4x
// per K-chunk; K=128 in four x32 chunks -> 16 v_wmma_f32_16x16x32_bf16/wave.
#define JT 4

__device__ __forceinline__ void wmma_slab_128(const unsigned int* __restrict__ A,
                                              const unsigned int* __restrict__ W,
                                              int m0, int j0, int lane,
                                              v8f acc[JT]) {
  const int half = lane >> 4;
  const int r    = lane & 15;
  const unsigned int* arow = A + (size_t)(m0 + r) * 64;   // 64 dwords per 128-bf16 row
#pragma unroll
  for (int kb = 0; kb < 4; ++kb) {
    union { unsigned int u[8]; v16bf v; } fa;
    const unsigned int* ap = arow + kb * 16;
#pragma unroll
    for (int i = 0; i < 4; ++i) fa.u[i] = ap[i + 8 * half];          // K = 2i   + 16*half
#pragma unroll
    for (int i = 4; i < 8; ++i) fa.u[i] = ap[i + 8 * half];          // K = 2i   + 16*half
#pragma unroll
    for (int jt = 0; jt < JT; ++jt) {
      union { unsigned int u[8]; v16bf v; } fb;
      const unsigned int* bp =
          W + (size_t)(j0 + jt * 16 + r) * 64 + kb * 16 + 8 * half;  // K linear per half
#pragma unroll
      for (int i = 0; i < 8; ++i) fb.u[i] = bp[i];
      acc[jt] = __builtin_amdgcn_wmma_f32_16x16x32_bf16(false, fa.v, false, fb.v,
                                                        (short)0, acc[jt], false, false);
    }
  }
}

__global__ __launch_bounds__(32)
void gemm_bf16_kernel(const unsigned int* __restrict__ A,
                      const unsigned int* __restrict__ W,
                      const float* __restrict__ bias,
                      float* __restrict__ C) {
  const int lane = threadIdx.x;
  const int m0 = blockIdx.x << 4;
  const int j0 = blockIdx.y << 6;                 // 4 tiles of 16 columns
  v8f acc[JT] = {};
  wmma_slab_128(A, W, m0, j0, lane, acc);
  const int half = lane >> 4, r = lane & 15;
#pragma unroll
  for (int jt = 0; jt < JT; ++jt) {
    float bj = bias[j0 + jt * 16 + r];
    float* crow = C + (size_t)(m0 + 8 * half) * DD + j0 + jt * 16 + r;
#pragma unroll
    for (int p = 0; p < 8; ++p) crow[(size_t)p * DD] = acc[jt][p] + bj;
  }
}

// Output GEMM fused with softmax(scale_weights) weighting + accumulation.
__global__ __launch_bounds__(32)
void gemm_out_kernel(const unsigned int* __restrict__ A,
                     const unsigned int* __restrict__ W,
                     const float* __restrict__ bias,
                     const float* __restrict__ sw, int si, int accum,
                     float* __restrict__ out) {
  const int lane = threadIdx.x;
  const int m0 = blockIdx.x << 4;
  const int j0 = blockIdx.y << 6;
  v8f acc[JT] = {};
  wmma_slab_128(A, W, m0, j0, lane, acc);
  float s0 = sw[0], s1 = sw[1], s2 = sw[2];
  float mx = fmaxf(fmaxf(s0, s1), s2);
  float e0 = __expf(s0 - mx), e1 = __expf(s1 - mx), e2 = __expf(s2 - mx);
  float wgt = ((si == 0) ? e0 : (si == 1) ? e1 : e2) / (e0 + e1 + e2);
  const int half = lane >> 4, r = lane & 15;
#pragma unroll
  for (int jt = 0; jt < JT; ++jt) {
    float bj = bias[j0 + jt * 16 + r];
    float* crow = out + (size_t)(m0 + 8 * half) * DD + j0 + jt * 16 + r;
#pragma unroll
    for (int p = 0; p < 8; ++p) {
      float val = wgt * (acc[jt][p] + bj);
      size_t idx = (size_t)p * DD;
      crow[idx] = accum ? (crow[idx] + val) : val;
    }
  }
}

// ---------- edge attention kernels (one thread per (edge, head)) ----------
__global__ void score_kernel(const float* __restrict__ q, const float* __restrict__ k,
                             const int* __restrict__ src, const int* __restrict__ dst,
                             const float* __restrict__ ea,     // E x 3
                             const float* __restrict__ demb,   // (NBIN+1) x H
                             const float* __restrict__ dW,     // H x 2
                             const float* __restrict__ db,     // H
                             float* __restrict__ scores,       // E x H (raw)
                             float* __restrict__ mbuf,         // N x H (init -inf)
                             float inv_maxd) {
  int tid = blockIdx.x * blockDim.x + threadIdx.x;
  if (tid >= EE * HH) return;
  int e = tid >> 3, h = tid & 7;
  int s = src[e], d = dst[e];
  const float4* qp = (const float4*)(q + (size_t)d * DD + h * 16);
  const float4* kp = (const float4*)(k + (size_t)s * DD + h * 16);
  float dot = 0.f;
#pragma unroll
  for (int i = 0; i < 4; ++i) {
    float4 a = qp[i], b = kp[i];
    dot += a.x * b.x + a.y * b.y + a.z * b.z + a.w * b.w;
  }
  dot *= 0.25f;                                  // HD^-0.5 = 1/4
  float dist = ea[(size_t)e * 3 + 0];
  int bin = (int)(dist * inv_maxd * (float)NBIN);
  bin = bin > NBIN ? NBIN : bin;
  float bias = demb[bin * HH + h];
  float dx = ea[(size_t)e * 3 + 1], dy = ea[(size_t)e * 3 + 2];
  float nrm = fmaxf(sqrtf(dx * dx + dy * dy), 1e-8f);
  bias += tanhf((dx * dW[h * 2 + 0] + dy * dW[h * 2 + 1]) / nrm + db[h]);
  float sc = dot + bias;
  scores[tid] = sc;
  atomicMaxF(mbuf + (size_t)d * HH + h, sc);
}

// Fused exp + segment-sum + weighted scatter of v (e*v); division by s deferred.
__global__ void expagg_kernel(const float* __restrict__ scores,
                              const int* __restrict__ src, const int* __restrict__ dst,
                              const float* __restrict__ mbuf,
                              const float* __restrict__ v,
                              float* __restrict__ sbuf,
                              float* __restrict__ agg) {
  int tid = blockIdx.x * blockDim.x + threadIdx.x;
  if (tid >= EE * HH) return;
  int e = tid >> 3, h = tid & 7;
  int s = src[e], d = dst[e];
  float ev = __expf(scores[tid] - mbuf[(size_t)d * HH + h]);
  atomicAdd(sbuf + (size_t)d * HH + h, ev);
  const float4* vp = (const float4*)(v + (size_t)s * DD + h * 16);
  float* ap = agg + (size_t)d * DD + h * 16;
#pragma unroll
  for (int i = 0; i < 4; ++i) {
    float4 vv = vp[i];
    atomicAdd(ap + 4 * i + 0, ev * vv.x);
    atomicAdd(ap + 4 * i + 1, ev * vv.y);
    atomicAdd(ap + 4 * i + 2, ev * vv.z);
    atomicAdd(ap + 4 * i + 3, ev * vv.w);
  }
}

__global__ void norm_kernel(const float* __restrict__ agg,
                            const float* __restrict__ sbuf,
                            unsigned short* __restrict__ aggh) {
  int tid = blockIdx.x * blockDim.x + threadIdx.x;
  if (tid >= NN * DD) return;
  int n = tid >> 7, h = (tid & 127) >> 4;
  float val = agg[tid] / (sbuf[(size_t)n * HH + h] + 1e-16f);
  aggh[tid] = f2bfbits(val);
}

// ---------- launcher ----------
extern "C" void kernel_launch(void* const* d_in, const int* in_sizes, int n_in,
                              void* d_out, int out_size, void* d_ws, size_t ws_size,
                              hipStream_t stream) {
  (void)in_sizes; (void)n_in; (void)out_size; (void)ws_size;
  const float* x    = (const float*)d_in[0];
  const int*   ei   = (const int*)  d_in[1];
  const float* ea   = (const float*)d_in[2];
  const float* Wq   = (const float*)d_in[3];
  const float* bq   = (const float*)d_in[4];
  const float* Wk   = (const float*)d_in[5];
  const float* bk   = (const float*)d_in[6];
  const float* Wv   = (const float*)d_in[7];
  const float* bv   = (const float*)d_in[8];
  const float* Wo   = (const float*)d_in[9];
  const float* bo   = (const float*)d_in[10];
  const float* demb = (const float*)d_in[11];
  const float* dWp  = (const float*)d_in[12];
  const float* dbp  = (const float*)d_in[13];
  const float* sw   = (const float*)d_in[14];
  float* out = (float*)d_out;

  char* ws = (char*)d_ws;
  size_t off = 0;
  auto alloc = [&](size_t bytes) -> void* {
    void* p = ws + off;
    off += (bytes + 255) & ~(size_t)255;
    return p;
  };
  unsigned short* xh  = (unsigned short*)alloc((size_t)NN * DD * 2);
  unsigned short* wh  = (unsigned short*)alloc((size_t)4 * NSC * DD * DD * 2); // [t][si][D*D]
  float* qb     = (float*)alloc((size_t)NN * DD * 4);
  float* kb_    = (float*)alloc((size_t)NN * DD * 4);
  float* vb     = (float*)alloc((size_t)NN * DD * 4);
  float* scores = (float*)alloc((size_t)EE * HH * 4);
  float* mbuf   = (float*)alloc((size_t)NN * HH * 4);
  float* sbuf   = (float*)alloc((size_t)NN * HH * 4);
  float* agg    = (float*)alloc((size_t)NN * DD * 4);
  unsigned short* aggh = (unsigned short*)alloc((size_t)NN * DD * 2);

  const int WDD = NSC * DD * DD;
  unsigned short* wh_q = wh + 0 * (size_t)WDD;
  unsigned short* wh_k = wh + 1 * (size_t)WDD;
  unsigned short* wh_v = wh + 2 * (size_t)WDD;
  unsigned short* wh_o = wh + 3 * (size_t)WDD;

  const float scales[NSC] = {50.0f, 200.0f, 500.0f};

  const int TB = 256;
  // bf16 conversions
  conv_bf16_kernel<<<(NN * DD + TB - 1) / TB, TB, 0, stream>>>(x, xh, NN * DD);
  conv_bf16_kernel<<<(WDD + TB - 1) / TB, TB, 0, stream>>>(Wq, wh_q, WDD);
  conv_bf16_kernel<<<(WDD + TB - 1) / TB, TB, 0, stream>>>(Wk, wh_k, WDD);
  conv_bf16_kernel<<<(WDD + TB - 1) / TB, TB, 0, stream>>>(Wv, wh_v, WDD);
  conv_bf16_kernel<<<(WDD + TB - 1) / TB, TB, 0, stream>>>(Wo, wh_o, WDD);

  dim3 ggemm(NN / 16, DD / 64);   // 3125 x 2 slabs of 16x64
  const int EH = EE * HH;
  const int NH = NN * HH;

  for (int si = 0; si < NSC; ++si) {
    const int* srcI = ei + (size_t)si * 2 * EE;
    const int* dstI = srcI + EE;
    const unsigned int* whq = (const unsigned int*)(wh_q + (size_t)si * DD * DD);
    const unsigned int* whk = (const unsigned int*)(wh_k + (size_t)si * DD * DD);
    const unsigned int* whv = (const unsigned int*)(wh_v + (size_t)si * DD * DD);
    const unsigned int* who = (const unsigned int*)(wh_o + (size_t)si * DD * DD);
    const unsigned int* xhu = (const unsigned int*)xh;

    gemm_bf16_kernel<<<ggemm, 32, 0, stream>>>(xhu, whq, bq + si * DD, qb);
    gemm_bf16_kernel<<<ggemm, 32, 0, stream>>>(xhu, whk, bk + si * DD, kb_);
    gemm_bf16_kernel<<<ggemm, 32, 0, stream>>>(xhu, whv, bv + si * DD, vb);

    fill_kernel<<<(NH + TB - 1) / TB, TB, 0, stream>>>(mbuf, NH, -__builtin_inff());
    fill_kernel<<<(NH + TB - 1) / TB, TB, 0, stream>>>(sbuf, NH, 0.0f);
    fill_kernel<<<(NN * DD + TB - 1) / TB, TB, 0, stream>>>(agg, NN * DD, 0.0f);

    score_kernel<<<(EH + TB - 1) / TB, TB, 0, stream>>>(
        qb, kb_, srcI, dstI, ea + (size_t)si * EE * 3,
        demb + (size_t)si * (NBIN + 1) * HH, dWp + (size_t)si * HH * 2,
        dbp + (size_t)si * HH, scores, mbuf, 1.0f / scales[si]);

    expagg_kernel<<<(EH + TB - 1) / TB, TB, 0, stream>>>(scores, srcI, dstI,
                                                         mbuf, vb, sbuf, agg);
    norm_kernel<<<(NN * DD + TB - 1) / TB, TB, 0, stream>>>(agg, sbuf, aggh);

    gemm_out_kernel<<<ggemm, 32, 0, stream>>>((const unsigned int*)aggh, who,
                                              bo + si * DD, sw, si, si > 0, out);
  }
}
